// Attention_52733608460319
// MI455X (gfx1250) — matrix operand order repaired and checked by reading, not compiled, not run on hardware
//
#include <hip/hip_runtime.h>

#define Bb   4
#define Nseq 1024
#define DIMM 1024
#define Hn   16
#define Dh   64
#define SWIN 256

typedef __bf16 bf16;
typedef bf16  v16bf __attribute__((ext_vector_type(16)));
typedef float v8f   __attribute__((ext_vector_type(8)));

// ---------------------------------------------------------------- convert
__global__ void f32_to_bf16_kernel(const float* __restrict__ in,
                                   bf16* __restrict__ out, int n) {
    int i = blockIdx.x * blockDim.x + threadIdx.x;
    if (i < n) out[i] = (bf16)in[i];
}

// async 16-byte global -> LDS copy (CDNA5 Tensor/Async path, ASYNCcnt-tracked)
__device__ __forceinline__ void async_copy_b128(unsigned lds_off, const bf16* gptr) {
    asm volatile("global_load_async_to_lds_b128 %0, %1, off"
                 :: "v"(lds_off), "v"((unsigned long long)(uintptr_t)gptr)
                 : "memory");
}
__device__ __forceinline__ void wait_async0() {
    asm volatile("s_wait_asynccnt 0x0" ::: "memory");
}

// ---------------------------------------------------------------- GEMM: C[M,N] = A[M,K] * W[N,K]^T (+bias)
// block tile 64(M) x 128(N); 8 waves; each wave: 1 A-frag reused over 4 WMMAs
__global__ __launch_bounds__(256)
void gemm_wmma_bf16(const bf16* __restrict__ A, const bf16* __restrict__ W,
                    const float* __restrict__ bias, float* __restrict__ C,
                    int M, int Nout, int K) {
    __shared__ bf16 At[64][40];
    __shared__ bf16 Wt[128][40];
    const int tid  = threadIdx.x;
    const int lane = tid & 31;
    const int wid  = tid >> 5;           // 8 waves
    const int mt   = wid & 3;            // 4 M-tiles of 16
    const int nq   = wid >> 2;           // 2 N-halves of 64
    const int m0 = blockIdx.y * 64, n0 = blockIdx.x * 128;
    const int arow = tid >> 2;           // 0..63  (A tile: 64x32, 8 bf16/thread)
    const int acg  = (tid & 3) * 8;      // 0,8,16,24
    const int wrow = tid >> 1;           // 0..127 (W tile: 128x32, 16 bf16/thread)
    const int wcg  = (tid & 1) * 16;     // 0,16
    const int fm = lane & 15, hi = lane >> 4;

    // wave-relative LDS byte offsets for async destinations
    const unsigned at_off = (unsigned)(uintptr_t)&At[arow][acg];
    const unsigned wt_off0 = (unsigned)(uintptr_t)&Wt[wrow][wcg];
    const unsigned wt_off1 = (unsigned)(uintptr_t)&Wt[wrow][wcg + 8];

    v8f acc[4] = {{}, {}, {}, {}};
    for (int k0 = 0; k0 < K; k0 += 32) {
        // async stage tiles straight into LDS (no VGPR round trip)
        async_copy_b128(at_off,  A + (size_t)(m0 + arow) * K + k0 + acg);
        async_copy_b128(wt_off0, W + (size_t)(n0 + wrow) * K + k0 + wcg);
        async_copy_b128(wt_off1, W + (size_t)(n0 + wrow) * K + k0 + wcg + 8);
        if (k0 + 32 < K) {  // global_prefetch_b8 for next K slice
            __builtin_prefetch(A + (size_t)(m0 + arow) * K + k0 + 32 + acg, 0, 1);
            __builtin_prefetch(W + (size_t)(n0 + wrow) * K + k0 + 32 + wcg, 0, 1);
        }
        wait_async0();
        __syncthreads();

        // A fragment 16x32: lane halves hold K 0..15 / 16..31
        v16bf a;
#pragma unroll
        for (int i = 0; i < 8; ++i) {
            a[i]     = At[mt * 16 + fm][hi * 16 + i];
            a[i + 8] = At[mt * 16 + fm][hi * 16 + 8 + i];
        }
        // 4 B fragments (32x16 each): lane = N col, halves hold K 0..15 / 16..31
#pragma unroll
        for (int j = 0; j < 4; ++j) {
            v16bf bfr;
#pragma unroll
            for (int i = 0; i < 16; ++i)
                bfr[i] = Wt[nq * 64 + j * 16 + fm][hi * 16 + i];
            acc[j] = __builtin_amdgcn_wmma_f32_16x16x32_bf16(false, a, false, bfr,
                                                             (short)0, acc[j],
                                                             false, false);
        }
        __syncthreads();
    }
#pragma unroll
    for (int r = 0; r < 8; ++r) {
        int gm = m0 + mt * 16 + r + 8 * hi;
#pragma unroll
        for (int j = 0; j < 4; ++j) {
            int gn = n0 + nq * 64 + j * 16 + fm;
            float bv = bias ? bias[gn] : 0.f;
            C[(size_t)gm * Nout + gn] = acc[j][r] + bv;
        }
    }
}

// ---------------------------------------------------------------- qk-LayerNorm + RoPE, split qkv -> [B,H,N,D] bf16
__global__ __launch_bounds__(256)
void ln_rope_kernel(const float* __restrict__ qkv,
                    const float* __restrict__ qn_w, const float* __restrict__ qn_b,
                    const float* __restrict__ kn_w, const float* __restrict__ kn_b,
                    bf16* __restrict__ qb, bf16* __restrict__ kb,
                    bf16* __restrict__ vb) {
    const int lane = threadIdx.x & 31;
    const int gw = (blockIdx.x * blockDim.x + threadIdx.x) >> 5;  // over B*H*N
    const int n = gw & (Nseq - 1);
    const int h = (gw >> 10) & (Hn - 1);
    const int b = gw >> 14;
    const size_t rowbase = (size_t)(b * Nseq + n) * (3 * Hn * Dh);
    const size_t obase   = ((size_t)(b * Hn + h) * Nseq + n) * Dh;

    // theta = n * 10000^(-2*lane/64); both elems (lane, lane+32) share it
    float theta = (float)n * __expf(-(float)(2 * lane) * (9.210340371976184f / 64.0f));
    float cs = __cosf(theta), sn = __sinf(theta);

#pragma unroll
    for (int s = 0; s < 2; ++s) {  // 0 = q, 1 = k
        const float* row = qkv + rowbase + (size_t)(s * Hn + h) * Dh;
        float x0 = row[lane], x1 = row[lane + 32];
        float sum = x0 + x1;
#pragma unroll
        for (int m = 16; m >= 1; m >>= 1) sum += __shfl_xor(sum, m);
        float mu = sum * (1.0f / 64.0f);
        float d0 = x0 - mu, d1 = x1 - mu;
        float vs = d0 * d0 + d1 * d1;
#pragma unroll
        for (int m = 16; m >= 1; m >>= 1) vs += __shfl_xor(vs, m);
        float inv = rsqrtf(vs * (1.0f / 64.0f) + 1e-5f);
        const float* wgt = (s == 0) ? qn_w : kn_w;
        const float* bia = (s == 0) ? qn_b : kn_b;
        float y0 = d0 * inv * wgt[lane] + bia[lane];
        float y1 = d1 * inv * wgt[lane + 32] + bia[lane + 32];
        float o0 = y0 * cs - y1 * sn;   // rotate_half low half
        float o1 = y1 * cs + y0 * sn;   // rotate_half high half
        bf16* out = (s == 0) ? qb : kb;
        out[obase + lane]      = (bf16)o0;
        out[obase + lane + 32] = (bf16)o1;
    }
    const float* vrow = qkv + rowbase + (size_t)(2 * Hn + h) * Dh;
    vb[obase + lane]      = (bf16)vrow[lane];
    vb[obase + lane + 32] = (bf16)vrow[lane + 32];
}

// ---------------------------------------------------------------- flash attention (sliding window + sink), one wave per (b,h,16-query block)
__global__ __launch_bounds__(256)
void attn_wmma_kernel(const bf16* __restrict__ qb, const bf16* __restrict__ kmat,
                      const bf16* __restrict__ vmat, const float* __restrict__ sinks,
                      bf16* __restrict__ attnb) {
    __shared__ bf16 pbuf[8][16][40];   // per-wave P-transpose staging
    const int lane = threadIdx.x & 31;
    const int wid  = threadIdx.x >> 5;
    const int gw   = blockIdx.x * 8 + wid;      // over B*H*(N/16) = 4096
    const int qblk = gw & 63;
    const int h    = (gw >> 6) & 15;
    const int b    = gw >> 10;
    const int q0   = qblk * 16;
    const size_t bh = (size_t)(b * Hn + h) * Nseq;
    const int fm = lane & 15, hi = lane >> 4;
    const float scale = 0.125f;  // 1/sqrt(64)

    // Q A-fragments, two 32-wide d-halves
    v16bf aq[2];
#pragma unroll
    for (int hd = 0; hd < 2; ++hd) {
        const bf16* qrow = qb + (bh + q0 + fm) * Dh + hd * 32;
#pragma unroll
        for (int i = 0; i < 8; ++i) {
            aq[hd][i]     = qrow[hi * 16 + i];
            aq[hd][i + 8] = qrow[hi * 16 + 8 + i];
        }
    }

    v8f o0 = {}, o1 = {}, o2 = {}, o3 = {};
    float rm[8], rs[8];
#pragma unroll
    for (int r = 0; r < 8; ++r) { rm[r] = -3.0e38f; rs[r] = 0.f; }

    int kbeg = q0 - (SWIN - 1);
    if (kbeg < 0) kbeg = 0;
    kbeg &= ~31;

    for (int kt = kbeg; kt <= q0 + 15; kt += 32) {
        // ---- scores for 32 keys (two 16-key subtiles), D=64 = 2 WMMA each
        v8f c0 = {}, c1 = {};
#pragma unroll
        for (int s = 0; s < 2; ++s) {
            int key  = kt + s * 16 + fm;
            int keyc = key < Nseq ? key : Nseq - 1;
            v8f cc = {};
#pragma unroll
            for (int hd = 0; hd < 2; ++hd) {
                v16bf bk;
                const bf16* krow = kmat + (bh + keyc) * Dh + hd * 32;
#pragma unroll
                for (int i = 0; i < 16; ++i) bk[i] = krow[hi * 16 + i];
                cc = __builtin_amdgcn_wmma_f32_16x16x32_bf16(false, aq[hd], false, bk,
                                                             (short)0, cc, false, false);
            }
            if (s == 0) c0 = cc; else c1 = cc;
        }
        // ---- masked online softmax (rows live in C-layout: row = r + 8*hi, col = fm)
        float p0[8], p1[8];
#pragma unroll
        for (int r = 0; r < 8; ++r) {
            int i  = q0 + r + 8 * hi;
            int j0 = kt + fm, j1 = kt + 16 + fm;
            bool a0 = (j0 <= i) && (j0 > i - SWIN);
            bool a1 = (j1 <= i) && (j1 > i - SWIN);
            float s0 = a0 ? c0[r] * scale : -3.0e38f;
            float s1 = a1 ? c1[r] * scale : -3.0e38f;
            float tm = fmaxf(s0, s1);
#pragma unroll
            for (int m = 8; m >= 1; m >>= 1) tm = fmaxf(tm, __shfl_xor(tm, m));
            float mnew = fmaxf(rm[r], tm);
            float resc = __expf(rm[r] - mnew);
            rm[r] = mnew;
            float e0 = a0 ? __expf(s0 - mnew) : 0.f;
            float e1 = a1 ? __expf(s1 - mnew) : 0.f;
            float ts = e0 + e1;
#pragma unroll
            for (int m = 8; m >= 1; m >>= 1) ts += __shfl_xor(ts, m);
            rs[r] = rs[r] * resc + ts;
            o0[r] *= resc; o1[r] *= resc; o2[r] *= resc; o3[r] *= resc;
            p0[r] = e0; p1[r] = e1;
        }
        // ---- transpose P (C-layout -> A-fragment) via LDS, explicit split-counter wait
#pragma unroll
        for (int r = 0; r < 8; ++r) {
            int row = r + 8 * hi;
            pbuf[wid][row][fm]      = (bf16)p0[r];
            pbuf[wid][row][fm + 16] = (bf16)p1[r];
        }
        asm volatile("s_wait_dscnt 0" ::: "memory");
        v16bf pa;
#pragma unroll
        for (int i = 0; i < 8; ++i) {
            pa[i]     = pbuf[wid][fm][hi * 16 + i];
            pa[i + 8] = pbuf[wid][fm][hi * 16 + 8 + i];
        }
        // ---- O += P * V, four 16-wide d-tiles
#pragma unroll
        for (int t = 0; t < 4; ++t) {
            v16bf bv;
            int dcol = t * 16 + fm;
#pragma unroll
            for (int i = 0; i < 16; ++i) {
                int key  = kt + hi * 16 + i;
                int keyc = key < Nseq ? key : Nseq - 1;
                bv[i] = vmat[(bh + keyc) * Dh + dcol];
            }
            v8f* op = (t == 0) ? &o0 : (t == 1) ? &o1 : (t == 2) ? &o2 : &o3;
            *op = __builtin_amdgcn_wmma_f32_16x16x32_bf16(false, pa, false, bv,
                                                          (short)0, *op, false, false);
        }
    }
    // ---- sink joins the denominator only; normalize and emit [B,N,H*D] bf16
    float sinkv = sinks[h];
#pragma unroll
    for (int r = 0; r < 8; ++r) {
        float l   = rs[r] + __expf(sinkv - rm[r]);
        float inv = 1.0f / l;
        int i = q0 + r + 8 * hi;
        size_t ob = ((size_t)(b * Nseq + i) * Hn + h) * Dh;
        attnb[ob + fm]      = (bf16)(o0[r] * inv);
        attnb[ob + 16 + fm] = (bf16)(o1[r] * inv);
        attnb[ob + 32 + fm] = (bf16)(o2[r] * inv);
        attnb[ob + 48 + fm] = (bf16)(o3[r] * inv);
    }
}

// ---------------------------------------------------------------- launch
extern "C" void kernel_launch(void* const* d_in, const int* in_sizes, int n_in,
                              void* d_out, int out_size, void* d_ws, size_t ws_size,
                              hipStream_t stream) {
    const float* x      = (const float*)d_in[0];
    const float* qkv_w  = (const float*)d_in[1];
    const float* qn_w   = (const float*)d_in[2];
    const float* qn_b   = (const float*)d_in[3];
    const float* kn_w   = (const float*)d_in[4];
    const float* kn_b   = (const float*)d_in[5];
    const float* sinks  = (const float*)d_in[6];
    const float* proj_w = (const float*)d_in[7];
    const float* proj_b = (const float*)d_in[8];
    float* out = (float*)d_out;

    char* ws = (char*)d_ws;
    bf16*  xb     = (bf16*)(ws);                          //  8 MiB  [4096,1024]
    bf16*  wqkvb  = (bf16*)(ws + (size_t)8  * (1 << 20)); //  6 MiB  [3072,1024]
    bf16*  wprojb = (bf16*)(ws + (size_t)14 * (1 << 20)); //  2 MiB  [1024,1024]
    float* qkvf   = (float*)(ws + (size_t)16 * (1 << 20));// 48 MiB  [4096,3072]
    bf16*  qbuf   = (bf16*)(ws + (size_t)64 * (1 << 20)); //  8 MiB  [B,H,N,D]
    bf16*  kbuf   = (bf16*)(ws + (size_t)72 * (1 << 20)); //  8 MiB
    bf16*  vbuf   = (bf16*)(ws + (size_t)80 * (1 << 20)); //  8 MiB
    bf16*  attnb  = (bf16*)(ws + (size_t)88 * (1 << 20)); //  8 MiB  [4096,1024]

    const int M = Bb * Nseq;  // 4096

    // 1) convert to bf16
    f32_to_bf16_kernel<<<(M * DIMM + 255) / 256, 256, 0, stream>>>(x, xb, M * DIMM);
    f32_to_bf16_kernel<<<(3 * Hn * Dh * DIMM + 255) / 256, 256, 0, stream>>>(
        qkv_w, wqkvb, 3 * Hn * Dh * DIMM);
    f32_to_bf16_kernel<<<(DIMM * Hn * Dh + 255) / 256, 256, 0, stream>>>(
        proj_w, wprojb, DIMM * Hn * Dh);

    // 2) QKV projection: [4096,3072] = xb @ wqkvb^T
    gemm_wmma_bf16<<<dim3(3072 / 128, M / 64), 256, 0, stream>>>(
        xb, wqkvb, nullptr, qkvf, M, 3072, DIMM);

    // 3) qk-norm + RoPE + split
    ln_rope_kernel<<<(Bb * Hn * Nseq * 32) / 256, 256, 0, stream>>>(
        qkvf, qn_w, qn_b, kn_w, kn_b, qbuf, kbuf, vbuf);

    // 4) sliding-window flash attention with sinks
    attn_wmma_kernel<<<(Bb * Hn * (Nseq / 16)) / 8, 256, 0, stream>>>(
        qbuf, kbuf, vbuf, sinks, attnb);

    // 5) output projection: [4096,1024] = attnb @ wprojb^T + proj_b
    gemm_wmma_bf16<<<dim3(DIMM / 128, M / 64), 256, 0, stream>>>(
        attnb, wprojb, proj_b, out, M, DIMM, DIMM);
}
